// FFNLayer_28793460752968
// MI455X (gfx1250) — matrix operand order repaired and checked by reading, not compiled, not run on hardware
//
#include <hip/hip_runtime.h>
#include <cstdint>
#include <cstddef>

// ---------------------------------------------------------------------------
// Tensor-ring FFN for MI455X (gfx1250, wave32).
//   B=64, L=96, D=512, F=2048, R=16, M = B*L = 6144.
// Big GEMMs run on v_wmma_f32_16x16x32_bf16 (f32 accumulate); small rank-16
// ring contractions stay exact f32 VALU.
// ---------------------------------------------------------------------------

typedef __attribute__((ext_vector_type(16))) __bf16 v16bf;
typedef __attribute__((ext_vector_type(8)))  __bf16 v8bf;
typedef __attribute__((ext_vector_type(8)))  float  v8f;

#define TR_B    64
#define TR_L    96
#define TR_D    512
#define TR_F    2048
#define TR_M    (TR_B * TR_L)    // 6144 rows

__device__ __forceinline__ unsigned short f2bf(float f) {
  unsigned u = __float_as_uint(f);
  u += 0x7fffu + ((u >> 16) & 1u);       // round-to-nearest-even
  return (unsigned short)(u >> 16);
}

// ---------------------------------------------------------------------------
// Fold five (16,2,16) cores + one (16,3,16) core into T[r0][l=96][r] (f32).
// Block 0 builds T1 (layer-1 input side), block 1 builds Tout (layer-2 out).
// LDS ping-pong; intermediates max 16*32*16 floats.
// ---------------------------------------------------------------------------
__global__ void tr_chain_kernel(const float* __restrict__ coresA2, const float* __restrict__ coreA3,
                                float* __restrict__ TA,
                                const float* __restrict__ coresB2, const float* __restrict__ coreB3,
                                float* __restrict__ TB) {
  const float* c2 = (blockIdx.x == 0) ? coresA2 : coresB2;   // 5 cores, 512 floats each
  const float* c3 = (blockIdx.x == 0) ? coreA3  : coreB3;    // (16,3,16)
  float*       T  = (blockIdx.x == 0) ? TA      : TB;        // [16][96][16]

  __shared__ float buf0[8192];
  __shared__ float buf1[4096];
  const int t = threadIdx.x;                                  // 256 threads

  for (int i = t; i < 512; i += 256) buf0[i] = c2[i];         // Cur = G1 [16][2][16]
  __syncthreads();

  int Mi = 2;
  float* cur = buf0;
  float* nxt = buf1;
  for (int k = 1; k < 5; ++k) {                               // fold G2..G5 (dim-2 cores)
    const float* G = c2 + k * 512;
    const int nout = 16 * (Mi * 2) * 16;
    for (int o = t; o < nout; o += 256) {
      const int tp   = o & 15;
      const int rest = o >> 4;
      const int mi   = rest % (Mi * 2);
      const int r0   = rest / (Mi * 2);
      const int m = mi >> 1, ii = mi & 1;
      float acc = 0.f;
      #pragma unroll
      for (int tt = 0; tt < 16; ++tt)
        acc += cur[(r0 * Mi + m) * 16 + tt] * G[(tt * 2 + ii) * 16 + tp];
      nxt[o] = acc;
    }
    __syncthreads();
    float* tmp = cur; cur = nxt; nxt = tmp;
    Mi *= 2;                                                  // ends at 32, cur == buf0
  }
  // final fold with the dim-3 core: T[(r0*96 + m*3+i)*16 + r]
  for (int o = t; o < 16 * 96 * 16; o += 256) {
    const int rr   = o & 15;
    const int rest = o >> 4;
    const int l    = rest % 96;
    const int r0   = rest / 96;
    const int m = l / 3, ii = l % 3;
    float acc = 0.f;
    #pragma unroll
    for (int tt = 0; tt < 16; ++tt)
      acc += cur[(r0 * 32 + m) * 16 + tt] * c3[(tt * 3 + ii) * 16 + rr];
    T[o] = acc;
  }
}

// ---------------------------------------------------------------------------
// Prep: x -> bf16 and permute the four GEMM B-operands into bf16 [N][K].
// blockIdx.y selects the task; grid-stride within task.
// ---------------------------------------------------------------------------
__global__ void prep_kernel(const float* __restrict__ x,
                            const float* __restrict__ c1_ind,    // [16][512][16]
                            const float* __restrict__ c1_off,    // [16][2048][16]
                            const float* __restrict__ c2_iff,    // [16][2048][16]
                            const float* __restrict__ c2_outd,   // [16][512][16]
                            unsigned short* __restrict__ xb,     // [6144][512]
                            unsigned short* __restrict__ W7t,    // [256][512]
                            unsigned short* __restrict__ W1t,    // [2048][256]
                            unsigned short* __restrict__ W2t,    // [256][2048]
                            unsigned short* __restrict__ W3t) {  // [512][256]
  const int task   = blockIdx.y;
  const int stride = gridDim.x * blockDim.x;
  const int tid0   = blockIdx.x * blockDim.x + threadIdx.x;
  if (task == 0) {
    for (int i = tid0; i < TR_M * TR_D; i += stride) xb[i] = f2bf(x[i]);
  } else if (task == 1) {          // W7t[n=r6*16+r7][k=d] = c1_ind[r6][d][r7]
    for (int i = tid0; i < 256 * 512; i += stride) {
      const int n = i >> 9, k = i & 511;
      W7t[i] = f2bf(c1_ind[((size_t)(n >> 4) * 512 + k) * 16 + (n & 15)]);
    }
  } else if (task == 2) {          // W1t[of][k=r*16+a] = c1_out_ff[r][of][a]
    for (int i = tid0; i < 2048 * 256; i += stride) {
      const int of = i >> 8, k = i & 255;
      W1t[i] = f2bf(c1_off[((size_t)(k >> 4) * 2048 + of) * 16 + (k & 15)]);
    }
  } else if (task == 3) {          // W2t[n=r1*16+r2][k=of] = c2_in_ff[r1][of][r2]
    for (int i = tid0; i < 256 * 2048; i += stride) {
      const int n = i >> 11, of = i & 2047;
      W2t[i] = f2bf(c2_iff[((size_t)(n >> 4) * 2048 + of) * 16 + (n & 15)]);
    }
  } else {                         // W3t[d][k=t6*16+a] = c2_outd[t6][d][a]
    for (int i = tid0; i < 512 * 256; i += stride) {
      const int d = i >> 8, k = i & 255;
      W3t[i] = f2bf(c2_outd[((size_t)(k >> 4) * 512 + d) * 16 + (k & 15)]);
    }
  }
}

// ---------------------------------------------------------------------------
// WMMA GEMM: C[M][N] = A[M][K] (bf16 row-major) x Bt[N][K] (bf16, N-major).
// One wave -> 16x64 output strip (A fragment reused across 4 WMMAs).
// Per-lane fragments use contiguous k (k = 16*half + s), same order for A and B:
//   A: lanes 0-15 row=lane, K 0..15; lanes 16-31 row=lane-16, K 16..31
//   B: lanes 0-15 col=lane, K 0..15; lanes 16-31 col=lane-16, K 16..31
//   C/D f32: VGPR i -> row i (lanes 0-15) / i+8 (lanes 16-31), col = lane&15
// Grid must give exactly (M/16)*(N/64) waves (EXEC all-ones for WMMA).
// ---------------------------------------------------------------------------
template<int OUT_BF16, int HAS_BIAS, int DO_RELU>
__global__ void wmma_gemm_kernel(const unsigned short* __restrict__ A,
                                 const unsigned short* __restrict__ Bt,
                                 float* __restrict__ Cf,
                                 unsigned short* __restrict__ Cb,
                                 const float* __restrict__ bias, int ldbias,
                                 int M, int N, int K) {
  const int wave = (blockIdx.x * blockDim.x + threadIdx.x) >> 5;
  const int lane = threadIdx.x & 31;
  const int ntg  = N >> 6;
  const int mt   = wave / ntg;
  const int ng   = wave - mt * ntg;
  const int half = lane >> 4;
  const int l16  = lane & 15;
  if (mt * 16 >= M) return;   // wave-uniform; never taken with exact grids

  const unsigned short* arow  = A  + (size_t)(mt * 16 + l16) * K + half * 16;
  const unsigned short* bcol0 = Bt + (size_t)(ng * 64 + l16) * K + half * 16;

  v8f acc[4];
  #pragma unroll
  for (int j = 0; j < 4; ++j)
    #pragma unroll
    for (int i = 0; i < 8; ++i) acc[j][i] = 0.f;

  for (int kk = 0; kk < K; kk += 32) {
    union { v16bf v; v8bf h[2]; } a;
    a.h[0] = *(const v8bf*)(arow + kk);
    a.h[1] = *(const v8bf*)(arow + kk + 8);
    #pragma unroll
    for (int j = 0; j < 4; ++j) {
      const v16bf b = *(const v16bf*)(bcol0 + (size_t)j * 16 * K + kk);
      acc[j] = __builtin_amdgcn_wmma_f32_16x16x32_bf16(
          false, a.v, false, b, (short)0, acc[j], false, false);
    }
  }

  #pragma unroll
  for (int j = 0; j < 4; ++j) {
    const int col = (ng * 4 + j) * 16 + l16;
    #pragma unroll
    for (int i = 0; i < 8; ++i) {
      const int row = mt * 16 + i + half * 8;
      float v = acc[j][i];
      if (HAS_BIAS) v += bias[(row % 96) * ldbias + col];   // rows are (b*96 + l)
      if (DO_RELU)  v = (v > 0.f) ? v : 0.f;
      const size_t idx = (size_t)row * N + col;
      if (OUT_BF16) Cb[idx] = f2bf(v);
      else          Cf[idx] = v;
    }
  }
}

// ---------------------------------------------------------------------------
// Ring closure (exact f32): s[b][r0*16+rr] = sum_{l,r} T[r0][l][r] * u[b*96+l][r*16+rr]
// ---------------------------------------------------------------------------
__global__ void ring_close_kernel(const float* __restrict__ T,   // [16][96][16]
                                  const float* __restrict__ u,   // [6144][256]
                                  float* __restrict__ s) {       // [64][256]
  const int b = blockIdx.x;
  const int t = threadIdx.x;
  const int r0 = t >> 4, rr = t & 15;
  float acc = 0.f;
  for (int l = 0; l < 96; ++l) {
    const float* urow = u + (size_t)(b * 96 + l) * 256;
    const float* Trow = T + (r0 * 96 + l) * 16;
    #pragma unroll
    for (int r = 0; r < 16; ++r)
      acc += Trow[r] * urow[r * 16 + rr];
  }
  s[b * 256 + t] = acc;
}

// ---------------------------------------------------------------------------
// Ring expansion (exact f32 -> bf16):
//   out[m][r*16+a] = sum_c s[b][a*16+c] * Tm[c][l][r],  m = b*96+l
// ---------------------------------------------------------------------------
__global__ void expand_kernel(const float* __restrict__ s,    // [64][256]  (a*16+c)
                              const float* __restrict__ Tm,   // [16][96][16]
                              unsigned short* __restrict__ out) { // [6144][256]
  const int m = blockIdx.x;
  const int b = m / 96, l = m % 96;
  __shared__ float sh[256];
  sh[threadIdx.x] = s[b * 256 + threadIdx.x];
  __syncthreads();
  const int r = threadIdx.x >> 4, a = threadIdx.x & 15;
  float acc = 0.f;
  #pragma unroll
  for (int c = 0; c < 16; ++c)
    acc += sh[a * 16 + c] * Tm[(c * 96 + l) * 16 + r];
  out[(size_t)m * 256 + threadIdx.x] = f2bf(acc);
}

// ---------------------------------------------------------------------------
extern "C" void kernel_launch(void* const* d_in, const int* in_sizes, int n_in,
                              void* d_out, int out_size, void* d_ws, size_t ws_size,
                              hipStream_t stream) {
  (void)in_sizes; (void)n_in; (void)out_size; (void)ws_size;
  const float* x       = (const float*)d_in[0];
  const float* c1_in2  = (const float*)d_in[1];   // (5,16,2,16)
  const float* c1_in3  = (const float*)d_in[2];   // (16,3,16)
  const float* c1_ind  = (const float*)d_in[3];   // (16,512,16)
  const float* c1_oseq = (const float*)d_in[4];   // (16,96,16)
  const float* c1_off  = (const float*)d_in[5];   // (16,2048,16)
  const float* b1      = (const float*)d_in[6];   // (96*2048)
  const float* c2_iseq = (const float*)d_in[7];   // (16,96,16)
  const float* c2_iff  = (const float*)d_in[8];   // (16,2048,16)
  const float* c2_out2 = (const float*)d_in[9];   // (5,16,2,16)
  const float* c2_out3 = (const float*)d_in[10];  // (16,3,16)
  const float* c2_outd = (const float*)d_in[11];  // (16,512,16)
  const float* b2      = (const float*)d_in[12];  // (96*512)

  // workspace layout (~54 MB, 256B aligned)
  char* ws = (char*)d_ws;
  size_t off = 0;
  auto alloc = [&](size_t bytes) -> char* {
    char* p = ws + off;
    off += (bytes + 255) & ~(size_t)255;
    return p;
  };
  unsigned short* xb  = (unsigned short*)alloc((size_t)TR_M * TR_D * 2);     // 6.3 MB
  unsigned short* W7t = (unsigned short*)alloc(256 * 512 * 2);
  unsigned short* W1t = (unsigned short*)alloc(2048 * 256 * 2);
  unsigned short* W2t = (unsigned short*)alloc(256 * 2048 * 2);
  unsigned short* W3t = (unsigned short*)alloc(512 * 256 * 2);
  float* T1   = (float*)alloc(16 * 96 * 16 * 4);
  float* Tout = (float*)alloc(16 * 96 * 16 * 4);
  float* u    = (float*)alloc((size_t)TR_M * 256 * 4);                        // 6.3 MB
  float* s1   = (float*)alloc(64 * 256 * 4);
  unsigned short* Pbf = (unsigned short*)alloc((size_t)TR_M * 256 * 2);       // 3.1 MB
  unsigned short* hbf = (unsigned short*)alloc((size_t)TR_M * TR_F * 2);      // 25 MB
  float* Q    = (float*)alloc((size_t)TR_M * 256 * 4);                        // 6.3 MB
  float* s2   = (float*)alloc(64 * 256 * 4);
  unsigned short* Ubf = (unsigned short*)alloc((size_t)TR_M * 256 * 2);       // 3.1 MB

  // 1) fold small cores into T1 (layer-1 in) and Tout (layer-2 out)
  tr_chain_kernel<<<2, 256, 0, stream>>>(c1_in2, c1_in3, T1, c2_out2, c2_out3, Tout);
  // 2) bf16 conversion + weight permutes
  prep_kernel<<<dim3(2048, 5), 256, 0, stream>>>(x, c1_ind, c1_off, c2_iff, c2_outd,
                                                 xb, W7t, W1t, W2t, W3t);
  // 3) u[m][r6*16+r7] = x[m][:] . c1_ind            (GEMM K=512, N=256)
  wmma_gemm_kernel<0, 0, 0><<<192, 256, 0, stream>>>(xb, W7t, u, nullptr, nullptr, 0,
                                                     TR_M, 256, TR_D);
  // 4) s1[b][r0,r7] = T1 . u  (exact)
  ring_close_kernel<<<64, 256, 0, stream>>>(T1, u, s1);
  // 5) P[m][r*16+a] = sum_c s1[b][a,c] * c1_out_seq[c][l][r]  (exact, -> bf16)
  expand_kernel<<<TR_M, 256, 0, stream>>>(s1, c1_oseq, Pbf);
  // 6) h = relu(P . H2^T + b1)                       (GEMM K=256, N=2048, bf16 out)
  wmma_gemm_kernel<1, 1, 1><<<1536, 256, 0, stream>>>(Pbf, W1t, nullptr, hbf, b1, TR_F,
                                                      TR_M, TR_F, 256);
  // 7) Q[m][r1*16+r2] = h . c2_in_ff^T               (GEMM K=2048, N=256)
  wmma_gemm_kernel<0, 0, 0><<<192, 256, 0, stream>>>(hbf, W2t, Q, nullptr, nullptr, 0,
                                                     TR_M, 256, TR_F);
  // 8) s2[b][r0,r2] = c2_in_seq . Q  (exact)
  ring_close_kernel<<<64, 256, 0, stream>>>(c2_iseq, Q, s2);
  // 9) U[m][t6*16+a] = sum_c s2[b][a,c] * Tout[c][l][t6]  (exact, -> bf16)
  expand_kernel<<<TR_M, 256, 0, stream>>>(s2, Tout, Ubf);
  // 10) y = U . c2_outd^T + b2                       (GEMM K=256, N=512, f32 out)
  wmma_gemm_kernel<0, 1, 0><<<384, 256, 0, stream>>>(Ubf, W3t, (float*)d_out, nullptr,
                                                     b2, TR_D, TR_M, TR_D, 256);
}
